// TypeCrossAttention_30537217474902
// MI455X (gfx1250) — matrix operand order repaired and checked by reading, not compiled, not run on hardware
//
#include <hip/hip_runtime.h>

// ---------------------------------------------------------------------------
// TypeCrossAttention for MI455X (gfx1250, wave32, WMMA f16->f32)
//
// Shapes: B=8, T1=128, T2=256, C=256, H=8, hd=32, E=32, A1=16, NTYPE=3
// Key identities: the aux_x1 contribution to K is a per-row softmax shift
// (drops out); the aux_x1 contribution to V factors out as +Vaux since
// softmax rows sum to 1. So this is 3 fused standard cross attentions.
// All matmuls: v_wmma_f32_16x16x32_f16. f16 tile staging uses the gfx1250
// async-to-LDS engine when the toolchain exposes it (ASYNCcnt tracked).
// ---------------------------------------------------------------------------

typedef _Float16 h8   __attribute__((ext_vector_type(8)));
typedef _Float16 v16h __attribute__((ext_vector_type(16)));
typedef float    v8f  __attribute__((ext_vector_type(8)));

#define LSTRIDE 40      // halves: 32-K staging tiles, padded (80B rows, 16B aligned)
#define PSTRIDE 264     // halves: P tile rows (528B, 16B aligned, conflict-free frags)
#define VSTRIDE 280     // halves: V^T rows (560B, 16B aligned, conflict-free frags)

#if defined(__HIP_DEVICE_COMPILE__) && \
    __has_builtin(__builtin_amdgcn_global_load_async_to_lds_b128) && \
    __has_builtin(__builtin_amdgcn_s_wait_asynccnt)
#define HAVE_ASYNC_LDS 1
#else
#define HAVE_ASYNC_LDS 0
#endif

#if HAVE_ASYNC_LDS
// b128 payload type per the compiler's diagnostic: int vector_size(16),
// global side in addrspace(1), LDS side in addrspace(3).
typedef int i4 __attribute__((vector_size(4 * sizeof(int))));
typedef __attribute__((address_space(1))) i4 gi4;
typedef __attribute__((address_space(3))) i4 li4;
// 16B global -> LDS copy through the async engine. Pointers are materialized
// via integer casts: generic LDS pointers carry the LDS offset in the low 32
// bits (CDNA5 aperture rule), and generic==global numerically for global mem.
__device__ __forceinline__ void async_ld16(const void* g, void* l) {
    __builtin_amdgcn_global_load_async_to_lds_b128(
        (gi4*)(unsigned long long)g,
        (li4*)(unsigned)(unsigned long long)l, 0, 0);
}
__device__ __forceinline__ void wait_async0() {
    __builtin_amdgcn_s_wait_asynccnt(0);
}
#endif

__device__ __forceinline__ v8f vzero() {
    v8f z; for (int i = 0; i < 8; ++i) z[i] = 0.0f; return z;
}

__device__ __forceinline__ v8f wmma32(v16h a, v16h b, v8f c) {
    // D = A(16x32 f16) x B(32x16 f16) + C(16x16 f32)
    return __builtin_amdgcn_wmma_f32_16x16x32_f16(false, a, false, b,
                                                  (short)0, c, false, false);
}

// Load one 16x32 f16 operand fragment from LDS (row-major, k contiguous).
// CDNA5 layout: lane l -> row l&15; lanes 0-15 carry K 0-7 & 16-23,
// lanes 16-31 carry K 8-15 & 24-31 (two 16B chunks per lane).
__device__ __forceinline__ v16h frag_ld(const _Float16* base, int stride_h) {
    const int lane = threadIdx.x & 31;
    const int row  = lane & 15;
    const int kb   = (lane >> 4) << 3;          // 0 or 8
    const _Float16* p = base + row * stride_h + kb;
    h8 lo = *(const h8*)(p);
    h8 hi = *(const h8*)(p + 16);
    return __builtin_shufflevector(lo, hi, 0,1,2,3,4,5,6,7,8,9,10,11,12,13,14,15);
}

// ---------------------------------------------------------------------------
// Generic projection GEMM: Out[t,b] (M x 256, f16) =
//      A1[b] (M x K1) @ W1[t] + A2[b] (M x K2<=16, zero-padded step) @ W2[t] + bias[t]
// Block: 128 thr (4 waves), tile 64x64, each wave a 32x32 quadrant (2x2 WMMA).
// A/W are f32 in global -> VALU-converted to f16 during staging (async engine
// can't convert, so this kernel keeps register staging + global_prefetch).
// ---------------------------------------------------------------------------
__global__ void __launch_bounds__(128)
proj_gemm_kernel(const float* __restrict__ A1, int rsA1, long long sA1b,
                 const float* __restrict__ A2, long long sA2b,
                 const float* __restrict__ W1, const float* __restrict__ W2,
                 long long sWt,
                 const float* __restrict__ bias,
                 _Float16* __restrict__ Out, long long sOutImg,
                 int M, int K1, int K2, int Bn)
{
    const int n0   = blockIdx.x * 64;
    const int m0   = blockIdx.y * 64;
    const int z    = blockIdx.z;
    const int b    = z % Bn, t = z / Bn;
    const int tid  = threadIdx.x;
    const int wave = tid >> 5, lane = tid & 31;

    __shared__ _Float16 lA[64 * LSTRIDE];
    __shared__ _Float16 lB[64 * LSTRIDE];

    const float* a1 = A1 ? A1 + (long long)b * sA1b : nullptr;
    const float* a2 = A2 ? A2 + (long long)b * sA2b : nullptr;
    const float* w1 = W1 ? W1 + (long long)t * sWt  : nullptr;
    const float* w2 = W2 ? W2 + (long long)t * sWt  : nullptr;

    v8f acc[2][2];
    acc[0][0] = vzero(); acc[0][1] = vzero();
    acc[1][0] = vzero(); acc[1][1] = vzero();

    const int steps1 = K1 >> 5;
    const int steps  = steps1 + (K2 > 0 ? 1 : 0);
    const int mw = (wave >> 1) << 5;      // 0 / 32
    const int nw = (wave & 1) << 5;       // 0 / 32

    for (int s = 0; s < steps; ++s) {
        const bool aux = (s >= steps1);
        // stage A tile 64x32 (f32 -> f16), zero-pad the aux K-step
        for (int idx = tid; idx < 64 * 32; idx += 128) {
            const int r = idx >> 5, k = idx & 31;
            float vfl = 0.0f;
            if (!aux)           vfl = a1[(long long)(m0 + r) * rsA1 + (s << 5) + k];
            else if (k < K2)    vfl = a2[(long long)(m0 + r) * K2 + k];
            lA[r * LSTRIDE + k] = (_Float16)vfl;
        }
        // stage W tile transposed: lB[n][k] = W[k_glob][n0+n]
        for (int idx = tid; idx < 64 * 32; idx += 128) {
            const int k = idx >> 6, n = idx & 63;
            float vfl = 0.0f;
            if (!aux)           vfl = w1[((long long)(s << 5) + k) * 256 + n0 + n];
            else if (k < K2)    vfl = w2[(long long)k * 256 + n0 + n];
            lB[n * LSTRIDE + k] = (_Float16)vfl;
        }
        __syncthreads();

        // prefetch next A tile (gfx1250 global_prefetch_b8)
        if (s + 1 < steps1) {
            const float* pf = a1 + (long long)(m0 + (tid >> 2)) * rsA1
                                 + ((s + 1) << 5) + ((tid & 3) << 3);
            __builtin_prefetch(pf, 0, 1);
        }

        v16h aF0 = frag_ld(lA + mw * LSTRIDE,        LSTRIDE);
        v16h aF1 = frag_ld(lA + (mw + 16) * LSTRIDE, LSTRIDE);
        v16h bF0 = frag_ld(lB + nw * LSTRIDE,        LSTRIDE);
        v16h bF1 = frag_ld(lB + (nw + 16) * LSTRIDE, LSTRIDE);
        acc[0][0] = wmma32(aF0, bF0, acc[0][0]);
        acc[0][1] = wmma32(aF0, bF1, acc[0][1]);
        acc[1][0] = wmma32(aF1, bF0, acc[1][0]);
        acc[1][1] = wmma32(aF1, bF1, acc[1][1]);
        __syncthreads();
    }

    // epilogue: + bias, cast f16, store (C layout: vgpr r -> M=r / M=8+r)
    _Float16* out = Out + (long long)z * sOutImg;
    const int lrow = (lane >> 4) << 3;
    const int lcol = lane & 15;
    for (int i = 0; i < 2; ++i)
        for (int j = 0; j < 2; ++j) {
            const int ncol = n0 + nw + j * 16 + lcol;
            const float bv = bias ? bias[(long long)t * 256 + ncol] : 0.0f;
            for (int r = 0; r < 8; ++r) {
                const int mrow = m0 + mw + i * 16 + lrow + r;
                out[(long long)mrow * 256 + ncol] = (_Float16)(acc[i][j][r] + bv);
            }
        }
}

// ---------------------------------------------------------------------------
// Attention: one WG per (type*B + b, head, 64-row t1 tile). Each wave owns a
// 16x256 S strip: 16 WMMAs for S=Q K^T (hd=32 -> one K-step), in-register
// masked softmax with half-wave __shfl_xor reductions, then P@V (16 WMMAs),
// + Vaux. Phase-aliased LDS (52 KB). Q/K tiles arrive via async-to-LDS.
// ---------------------------------------------------------------------------
__global__ void __launch_bounds__(128)
attn_kernel(const _Float16* __restrict__ Q, const _Float16* __restrict__ K,
            const _Float16* __restrict__ V, const _Float16* __restrict__ Vaux,
            const int* __restrict__ masks, _Float16* __restrict__ Y, int Bn)
{
    const int m0   = blockIdx.x * 64;
    const int h    = blockIdx.y;
    const int z    = blockIdx.z;
    const int tid  = threadIdx.x;
    const int wave = tid >> 5, lane = tid & 31;

    __shared__ __align__(16) char smem[64 * PSTRIDE * 2 + 32 * VSTRIDE * 2]; // 51712 B
    _Float16* lP = (_Float16*)smem;                            // phase 2
    _Float16* lV = (_Float16*)(smem + 64 * PSTRIDE * 2);       // phase 2
    _Float16* lQ = (_Float16*)smem;                            // phase 1 (aliased)
    _Float16* lK = (_Float16*)(smem + 64 * LSTRIDE * 2);       // phase 1 (aliased)

    const _Float16* Qi  = Q    + (long long)z * 128 * 256;
    const _Float16* Ki  = K    + (long long)z * 256 * 256;
    const _Float16* Vi  = V    + (long long)z * 256 * 256;
    const _Float16* Vxi = Vaux + (long long)z * 128 * 256;
    const int*      Mi  = masks + (long long)z * 128 * 256;
    _Float16*       Yi  = Y    + (long long)z * 128 * 256;

    // ---- phase 1: stage head slices of Q (64x32) and K (256x32)
#if HAVE_ASYNC_LDS
    for (int idx = tid; idx < 64 * 4; idx += 128) {
        const int r = idx >> 2, half = idx & 3;
        async_ld16(Qi + (long long)(m0 + r) * 256 + h * 32 + half * 8,
                   lQ + r * LSTRIDE + half * 8);
    }
    for (int idx = tid; idx < 256 * 4; idx += 128) {
        const int r = idx >> 2, half = idx & 3;
        async_ld16(Ki + (long long)r * 256 + h * 32 + half * 8,
                   lK + r * LSTRIDE + half * 8);
    }
    wait_async0();
#else
    for (int idx = tid; idx < 64 * 4; idx += 128) {
        const int r = idx >> 2, half = idx & 3;
        *(h8*)(lQ + r * LSTRIDE + half * 8) =
            *(const h8*)(Qi + (long long)(m0 + r) * 256 + h * 32 + half * 8);
    }
    for (int idx = tid; idx < 256 * 4; idx += 128) {
        const int r = idx >> 2, half = idx & 3;
        *(h8*)(lK + r * LSTRIDE + half * 8) =
            *(const h8*)(Ki + (long long)r * 256 + h * 32 + half * 8);
    }
#endif
    __syncthreads();

    // ---- S = Q K^T for this wave's 16-row strip (full 256 cols)
    v8f accS[16];
    {
        const v16h qF = frag_ld(lQ + (wave * 16) * LSTRIDE, LSTRIDE);
        for (int j = 0; j < 16; ++j) {
            v16h kF = frag_ld(lK + (j * 16) * LSTRIDE, LSTRIDE);
            v8f  c  = vzero();
            accS[j] = wmma32(qF, kF, c);
        }
    }

    // ---- masked softmax (row = t1, cols = t2)
    const float scale = 0.17677669529663687f;    // 1/sqrt(32)
    const int lrow = (lane >> 4) << 3;
    const int lcol = lane & 15;
    float pm[8], anyv[8], rs[8];
    for (int r = 0; r < 8; ++r) { pm[r] = -3.0e38f; anyv[r] = 0.0f; rs[r] = 0.0f; }

    for (int j = 0; j < 16; ++j) {
        const int t2 = j * 16 + lcol;
        for (int r = 0; r < 8; ++r) {
            const int t1 = m0 + wave * 16 + lrow + r;
            const int mk = Mi[(long long)t1 * 256 + t2];
            float sv = accS[j][r] * scale;
            if (mk == 0) sv = -3.0e38f; else anyv[r] = 1.0f;
            accS[j][r] = sv;
            pm[r] = fmaxf(pm[r], sv);
        }
    }
    for (int r = 0; r < 8; ++r) {                 // reduce over the 16 lanes of a half
        float m_ = pm[r], a_ = anyv[r];
        for (int sh = 1; sh < 16; sh <<= 1) {
            m_ = fmaxf(m_, __shfl_xor(m_, sh, 32));
            a_ = fmaxf(a_, __shfl_xor(a_, sh, 32));
        }
        pm[r] = m_; anyv[r] = a_;
    }
    for (int j = 0; j < 16; ++j)
        for (int r = 0; r < 8; ++r) {
            // masked entries: exp(-3e38 - m) underflows to exactly 0
            const float p = (anyv[r] > 0.0f) ? __expf(accS[j][r] - pm[r]) : 1.0f;
            accS[j][r] = p;
            rs[r] += p;
        }
    for (int r = 0; r < 8; ++r) {
        float s_ = rs[r];
        for (int sh = 1; sh < 16; sh <<= 1) s_ += __shfl_xor(s_, sh, 32);
        rs[r] = 1.0f / s_;                        // all-masked rows: 1/256 uniform
    }
    __syncthreads();   // everyone done reading lQ/lK before aliased overwrite

    // ---- phase 2: stage V^T (lV[d][t2]) and write normalized P to LDS
    for (int idx = tid; idx < 32 * 256; idx += 128) {
        const int d = idx & 31, t2 = idx >> 5;
        lV[d * VSTRIDE + t2] = Vi[(long long)t2 * 256 + h * 32 + d];
    }
    for (int j = 0; j < 16; ++j) {
        const int t2 = j * 16 + lcol;
        for (int r = 0; r < 8; ++r) {
            const int rl = wave * 16 + lrow + r;
            lP[rl * PSTRIDE + t2] = (_Float16)(accS[j][r] * rs[r]);
        }
    }
    __syncthreads();

    // ---- Y = P @ V  (16 rows x 32 cols, 8 K-steps of 32 over t2)
    v8f accY[2];
    accY[0] = vzero(); accY[1] = vzero();
    for (int s = 0; s < 8; ++s) {
        v16h pF  = frag_ld(lP + (wave * 16) * PSTRIDE + s * 32, PSTRIDE);
        v16h vF0 = frag_ld(lV + 0  * VSTRIDE + s * 32, VSTRIDE);
        v16h vF1 = frag_ld(lV + 16 * VSTRIDE + s * 32, VSTRIDE);
        accY[0] = wmma32(pF, vF0, accY[0]);
        accY[1] = wmma32(pF, vF1, accY[1]);
    }
    for (int nt = 0; nt < 2; ++nt) {
        const int d = nt * 16 + lcol;
        for (int r = 0; r < 8; ++r) {
            const int t1 = m0 + wave * 16 + lrow + r;
            const float yv = accY[nt][r] + (float)Vxi[(long long)t1 * 256 + h * 32 + d];
            Yi[(long long)t1 * 256 + h * 32 + d] = (_Float16)yv;
        }
    }
}

// ---------------------------------------------------------------------------
// Output projection: out[b] = sum_t ( Y[t,b] @ Wp[t] + bp[t] )   (f32 out)
// Same 64x64 tiling; 3 types x 8 K-steps = 24 WMMA rounds per quadrant.
// Y tiles (already f16) arrive via async-to-LDS; Wp needs f32->f16 VALU.
// ---------------------------------------------------------------------------
__global__ void __launch_bounds__(128)
outproj_kernel(const _Float16* __restrict__ Y, const float* __restrict__ Wp,
               const float* __restrict__ bp, float* __restrict__ Outp, int Bn)
{
    const int n0   = blockIdx.x * 64;
    const int m0   = blockIdx.y * 64;
    const int b    = blockIdx.z;
    const int tid  = threadIdx.x;
    const int wave = tid >> 5, lane = tid & 31;

    __shared__ _Float16 lA[64 * LSTRIDE];
    __shared__ _Float16 lB[64 * LSTRIDE];

    v8f acc[2][2];
    acc[0][0] = vzero(); acc[0][1] = vzero();
    acc[1][0] = vzero(); acc[1][1] = vzero();
    const int mw = (wave >> 1) << 5;
    const int nw = (wave & 1) << 5;

    for (int t = 0; t < 3; ++t) {
        const _Float16* Yi = Y  + ((long long)t * Bn + b) * 128 * 256;
        const float*    w  = Wp + (long long)t * 256 * 256;
        for (int s = 0; s < 8; ++s) {
#if HAVE_ASYNC_LDS
            for (int idx = tid; idx < 64 * 4; idx += 128) {
                const int r = idx >> 2, half = idx & 3;
                async_ld16(Yi + (long long)(m0 + r) * 256 + s * 32 + half * 8,
                           lA + r * LSTRIDE + half * 8);
            }
#else
            for (int idx = tid; idx < 64 * 4; idx += 128) {
                const int r = idx >> 2, half = idx & 3;
                *(h8*)(lA + r * LSTRIDE + half * 8) =
                    *(const h8*)(Yi + (long long)(m0 + r) * 256 + s * 32 + half * 8);
            }
#endif
            for (int idx = tid; idx < 64 * 32; idx += 128) {
                const int k = idx >> 6, n = idx & 63;
                lB[n * LSTRIDE + k] =
                    (_Float16)w[((long long)s * 32 + k) * 256 + n0 + n];
            }
#if HAVE_ASYNC_LDS
            wait_async0();
#endif
            __syncthreads();
            v16h aF0 = frag_ld(lA + mw * LSTRIDE,        LSTRIDE);
            v16h aF1 = frag_ld(lA + (mw + 16) * LSTRIDE, LSTRIDE);
            v16h bF0 = frag_ld(lB + nw * LSTRIDE,        LSTRIDE);
            v16h bF1 = frag_ld(lB + (nw + 16) * LSTRIDE, LSTRIDE);
            acc[0][0] = wmma32(aF0, bF0, acc[0][0]);
            acc[0][1] = wmma32(aF0, bF1, acc[0][1]);
            acc[1][0] = wmma32(aF1, bF0, acc[1][0]);
            acc[1][1] = wmma32(aF1, bF1, acc[1][1]);
            __syncthreads();
        }
    }

    float* out = Outp + (long long)b * 128 * 256;
    const int lrow = (lane >> 4) << 3;
    const int lcol = lane & 15;
    for (int i = 0; i < 2; ++i)
        for (int j = 0; j < 2; ++j) {
            const int ncol = n0 + nw + j * 16 + lcol;
            const float bsum = bp[ncol] + bp[256 + ncol] + bp[512 + ncol];
            for (int r = 0; r < 8; ++r) {
                const int mrow = m0 + mw + i * 16 + lrow + r;
                out[(long long)mrow * 256 + ncol] = acc[i][j][r] + bsum;
            }
        }
}

// ---------------------------------------------------------------------------
extern "C" void kernel_launch(void* const* d_in, const int* in_sizes, int n_in,
                              void* d_out, int out_size, void* d_ws, size_t ws_size,
                              hipStream_t stream)
{
    const float* x1     = (const float*)d_in[0];
    const float* x2     = (const float*)d_in[1];
    const float* aux_x1 = (const float*)d_in[2];
    const float* aux_x2 = (const float*)d_in[3];
    const int*   masks  = (const int*)  d_in[4];
    const float* Wq     = (const float*)d_in[5];
    const float* bq     = (const float*)d_in[6];
    const float* Wk     = (const float*)d_in[7];
    const float* bk     = (const float*)d_in[8];
    const float* Wv     = (const float*)d_in[9];
    const float* bv     = (const float*)d_in[10];
    const float* Wp     = (const float*)d_in[11];
    const float* bp     = (const float*)d_in[12];
    float* out = (float*)d_out;

    const int B = 8;
    const long long Qn  = 3LL * 8 * 128 * 256;   // per-array f16 element counts
    const long long Kn  = 3LL * 8 * 256 * 256;
    _Float16* wsQ    = (_Float16*)d_ws;
    _Float16* wsK    = wsQ    + Qn;
    _Float16* wsV    = wsK    + Kn;
    _Float16* wsVaux = wsV    + Kn;
    _Float16* wsY    = wsVaux + Qn;

    // Q = x1 @ Wq + bq                       (M=128, K=256)
    proj_gemm_kernel<<<dim3(4, 2, 24), 128, 0, stream>>>(
        x1, 256, 128LL * 256, nullptr, 0,
        Wq, nullptr, 256LL * 256, bq, wsQ, 128LL * 256, 128, 256, 0, B);
    // K = x2 @ Wk[0:256] + aux_x2 @ Wk[272:288] + bk   (M=256, K=256+16)
    proj_gemm_kernel<<<dim3(4, 4, 24), 128, 0, stream>>>(
        x2, 256, 256LL * 256, aux_x2, 256LL * 16,
        Wk, Wk + 272LL * 256, 288LL * 256, bk, wsK, 256LL * 256, 256, 256, 16, B);
    // V = x2 @ Wv[0:256] + aux_x2 @ Wv[272:288] + bv
    proj_gemm_kernel<<<dim3(4, 4, 24), 128, 0, stream>>>(
        x2, 256, 256LL * 256, aux_x2, 256LL * 16,
        Wv, Wv + 272LL * 256, 288LL * 256, bv, wsV, 256LL * 256, 256, 256, 16, B);
    // Vaux = aux_x1 @ Wv[256:272]            (M=128, K=16, no bias)
    proj_gemm_kernel<<<dim3(4, 2, 24), 128, 0, stream>>>(
        nullptr, 0, 0, aux_x1, 128LL * 16,
        nullptr, Wv + 256LL * 256, 288LL * 256, nullptr, wsVaux, 128LL * 256,
        128, 0, 16, B);
    // attention per (type,b,head,t1-tile)
    attn_kernel<<<dim3(2, 8, 24), 128, 0, stream>>>(
        wsQ, wsK, wsV, wsVaux, masks, wsY, B);
    // out = sum_t Y_t @ Wp_t + bp_t
    outproj_kernel<<<dim3(4, 2, 8), 128, 0, stream>>>(wsY, Wp, bp, out, B);
}
